// DeformConv_50714973831354
// MI455X (gfx1250) — matrix operand order repaired and checked by reading, not compiled, not run on hardware
//
#include <hip/hip_runtime.h>
#include <hip/hip_bf16.h>
#include <math.h>
#include <stdint.h>

typedef _Float16 v8h  __attribute__((ext_vector_type(8)));
typedef _Float16 v16h __attribute__((ext_vector_type(16)));
typedef float    v8f  __attribute__((ext_vector_type(8)));
typedef unsigned v4u  __attribute__((ext_vector_type(4)));
typedef int      v4i  __attribute__((ext_vector_type(4)));
typedef int      v8i  __attribute__((ext_vector_type(8)));

#define HH     96
#define WW_    96
#define HW     9216      // 96*96
#define CI_    256
#define CO_    256
#define KK_    9
#define KDIM   2304      // KK_*CI_
#define NB     4
#define NCHUNK 72        // KDIM/32

#if defined(__HIP_DEVICE_COMPILE__) && __has_builtin(__builtin_amdgcn_tensor_load_to_lds) && __has_builtin(__builtin_amdgcn_s_wait_tensorcnt)
#define USE_TDM 1
#else
#define USE_TDM 0
#endif

__device__ __forceinline__ v16h cat16(v8h lo, v8h hi) {
  return __builtin_shufflevector(lo, hi, 0,1,2,3,4,5,6,7,8,9,10,11,12,13,14,15);
}

#if USE_TDM
// ---------------------------------------------------------------------------
// TDM 2D tile load: 'rows' rows of 32 f16 (64 B), row stride KDIM f16, from
// gsrc into LDS byte offset ldsOff.  D# per CDNA5 ISA ch.8:
//   group0: [1:0]=count=1, [63:32]=lds_addr, [120:64]=global_addr,
//           [127:126]=type=2
//   group1: [17:16]=data_size(2B), [79:48]=tensor_dim0, [111:80]=tensor_dim1,
//           [127:112]=tile_dim0, [143:128]=tile_dim1, [207:160]=dim0_stride
// Groups 2/3 unused (2D tensor).  6-arg builtin form (clang-23 lane).
// ---------------------------------------------------------------------------
__device__ __forceinline__ void tdm_load_w_chunk(unsigned ldsOff,
                                                 const _Float16* gsrc,
                                                 int rows) {
  unsigned long long ga = (unsigned long long)(uintptr_t)gsrc;
  v4u g0 = {0u, 0u, 0u, 0u};
  g0.x = 1u;                                         // count = 1, user mode
  g0.y = ldsOff;                                     // LDS byte address
  g0.z = (unsigned)ga;                               // global_addr[31:0]
  g0.w = (unsigned)((ga >> 32) & 0x1ffffffu)         // global_addr[56:32]
       | 0x80000000u;                                // type = 2 ("image")
  v8i g1 = {0, 0, 0, 0, 0, 0, 0, 0};
  g1.s0 = 0x00010000;                                // data_size = 2 bytes
  g1.s1 = (int)(((unsigned)KDIM & 0xffffu) << 16);   // tensor_dim0[15:0]
  g1.s2 = (int)(((unsigned)rows) << 16);             // tensor_dim1[15:0]
  g1.s3 = (int)(32u << 16);                          // tile_dim0 = 32
  g1.s4 = rows;                                      // tile_dim1 = rows
  g1.s5 = KDIM;                                      // tensor_dim0_stride
  v4i gz4 = {0, 0, 0, 0};
  v8i gz8 = {0, 0, 0, 0, 0, 0, 0, 0};
  __builtin_amdgcn_tensor_load_to_lds(g0, g1, gz4, gz4, gz8, 0);
}
#endif

// ---------------------------------------------------------------------------
// Repack weights to f16, K-major (k = kk*256 + c), fragment-friendly.
// ---------------------------------------------------------------------------
__global__ __launch_bounds__(256) void k_cvt_weights(
    const float* __restrict__ w, const float* __restrict__ w_off,
    _Float16* __restrict__ Whf, _Float16* __restrict__ Wohf)
{
  int idx = blockIdx.x * 256 + threadIdx.x;
  if (idx < CO_ * KDIM) {
    int o = idx / KDIM, k = idx % KDIM;
    int kk = k >> 8, c = k & 255;
    Whf[idx] = (_Float16)w[(o * CI_ + c) * KK_ + kk];
  } else {
    int j = idx - CO_ * KDIM;           // < 32*KDIM by grid sizing
    int n = j / KDIM, k = j % KDIM;
    int kk = k >> 8, c = k & 255;
    Wohf[j] = (n < 27) ? (_Float16)w_off[(n * CI_ + c) * KK_ + kk]
                       : (_Float16)0.0f;
  }
}

// ---------------------------------------------------------------------------
// Offset conv: implicit GEMM, M = 32 (27 padded) channels x N = 64 positions
// per WG, K = 2304 in chunks of 32.  Weight panel staged by TDM (dbl-buffer).
// ---------------------------------------------------------------------------
__global__ __launch_bounds__(256) void k_offset_conv(
    const float* __restrict__ x, const _Float16* __restrict__ Wohf,
    const float* __restrict__ b_off, float* __restrict__ om)
{
  __shared__ _Float16 Wl[2][32 * 32];   // weight chunk (A), 2 x 2 KB
  __shared__ _Float16 S[64 * 32];       // im2col chunk (B), 4 KB

  int t  = threadIdx.x;
  int p0 = blockIdx.x * 64;             // 576 WGs, tiles never cross b
  int b  = p0 / HW, s0 = p0 % HW;
  const float* xb = x + b * CI_ * HW;

  int lane = t & 31, wv = t >> 5;
  int nt = wv & 3;                      // position subtile (4)
  int mt = wv >> 2;                     // channel  subtile (2)
  int half = lane >> 4, ln = lane & 15;

  int m  = t >> 2;                      // producer: position 0..63
  int cq = (t & 3) * 8;                 // producer: 8 channels within chunk
  int s  = s0 + m;
  int hy = s / WW_, wxp = s % WW_;

  v8f acc = {};

#if USE_TDM
  if (wv == 0)
    tdm_load_w_chunk((unsigned)(uintptr_t)&Wl[0][0], Wohf, 32);
#endif

  for (int kc = 0; kc < NCHUNK; ++kc) {
    int kk = kc >> 3, c0 = (kc & 7) << 5;
    int dh = kk / 3 - 1, dw = kk % 3 - 1;
    int buf = kc & 1;

#if !USE_TDM
    v8h wreg = {};
    if (t < 128)
      wreg = *(const v8h*)(Wohf + (t >> 2) * KDIM + kc * 32 + (t & 3) * 8);
#endif

    v8h vals;
    {
      int hs = hy + dh, ws = wxp + dw;
      bool in = (hs >= 0) & (hs < HH) & (ws >= 0) & (ws < WW_);
      int hsc = min(max(hs, 0), HH - 1);
      int wsc = min(max(ws, 0), WW_ - 1);
      float fin = in ? 1.0f : 0.0f;
      const float* px = xb + (c0 + cq) * HW + hsc * WW_ + wsc;
      #pragma unroll
      for (int i = 0; i < 8; ++i)
        vals[i] = (_Float16)(fin * px[i * HW]);
    }

    __syncthreads();                    // prior chunk's fragment reads done
#if USE_TDM
    if (wv == 0 && kc + 1 < NCHUNK)
      tdm_load_w_chunk((unsigned)(uintptr_t)&Wl[buf ^ 1][0],
                       Wohf + (kc + 1) * 32, 32);
#else
    if (t < 128)
      *(v8h*)(&Wl[0][0] + (t >> 2) * 32 + (t & 3) * 8) = wreg;
    buf = 0;
#endif
    *(v8h*)(S + m * 32 + cq) = vals;
#if USE_TDM
    if (wv == 0) {                      // in-order TDM completion per wave
      if (kc + 1 < NCHUNK) __builtin_amdgcn_s_wait_tensorcnt(1);
      else                 __builtin_amdgcn_s_wait_tensorcnt(0);
    }
#endif
    __syncthreads();

    v8h blo = *(const v8h*)(S + (nt * 16 + ln) * 32 + half * 8);
    v8h bhi = *(const v8h*)(S + (nt * 16 + ln) * 32 + 16 + half * 8);
    v16h bf = cat16(blo, bhi);
    int ch = mt * 16 + ln;
    v8h alo = *(const v8h*)(&Wl[buf][0] + ch * 32 + half * 8);
    v8h ahi = *(const v8h*)(&Wl[buf][0] + ch * 32 + 16 + half * 8);
    v16h af = cat16(alo, ahi);
    acc = __builtin_amdgcn_wmma_f32_16x16x32_f16(false, af, false, bf,
                                                 (short)0, acc, false, false);
  }

  int spos = s0 + nt * 16 + ln;
  #pragma unroll
  for (int r = 0; r < 8; ++r) {
    int ch = mt * 16 + half * 8 + r;
    if (ch < 27)
      om[(b * 27 + ch) * HW + spos] = acc[r] + b_off[ch];
  }
}

// ---------------------------------------------------------------------------
// Per (b,kk,h,w): bilinear corner weights (x sigmoid(mask), validity folded
// in) + packed clamped corner coords.
// ---------------------------------------------------------------------------
__global__ __launch_bounds__(256) void k_prep(
    const float* __restrict__ om, float4* __restrict__ descW,
    unsigned* __restrict__ descC)
{
  int d  = blockIdx.x * 256 + threadIdx.x;   // NB*KK_*HW = 331776 exactly
  int sp = d % HW;
  int kk = (d / HW) % KK_;
  int b  = d / (KK_ * HW);
  int hq = sp / WW_, wq = sp % WW_;
  const float* omb = om + b * 27 * HW;

  float oh = omb[kk * HW + sp];
  float ow = omb[(KK_ + kk) * HW + sp];
  float mk = omb[(2 * KK_ + kk) * HW + sp];
  float mv = 1.0f / (1.0f + __expf(-mk));

  float hs = oh + (float)hq + (float)(kk / 3 - 1);
  float ws = ow + (float)wq + (float)(kk % 3 - 1);
  bool valid = (hs > -1.0f) & (ws > -1.0f) & (hs < (float)HH) & (ws < (float)WW_);

  float h0f = floorf(hs), w0f = floorf(ws);
  int h0 = (int)h0f, w0 = (int)w0f;
  float lh = hs - h0f, lw = ws - w0f;
  float hh = 1.0f - lh, hw = 1.0f - lw;

  float w00 = (valid && h0 >= 0        && w0 >= 0        ) ? hh * hw : 0.0f;
  float w01 = (valid && h0 >= 0        && w0 + 1 <= WW_-1) ? hh * lw : 0.0f;
  float w10 = (valid && h0 + 1 <= HH-1 && w0 >= 0        ) ? lh * hw : 0.0f;
  float w11 = (valid && h0 + 1 <= HH-1 && w0 + 1 <= WW_-1) ? lh * lw : 0.0f;

  int h0c = min(max(h0, 0), HH - 1),  h1c = min(max(h0 + 1, 0), HH - 1);
  int w0c = min(max(w0, 0), WW_ - 1), w1c = min(max(w0 + 1, 0), WW_ - 1);

  descW[d] = make_float4(w00 * mv, w01 * mv, w10 * mv, w11 * mv);
  descC[d] = (unsigned)h0c | ((unsigned)h1c << 8) |
             ((unsigned)w0c << 16) | ((unsigned)w1c << 24);
}

// ---------------------------------------------------------------------------
// Fused sampler + main GEMM.  WG tile: 256 channels x 64 positions, K chunks
// of 32.  Weight panel (16 KB/chunk) staged by TDM, double-buffered so the
// DMA overlaps the bilinear gathers and the 8 back-to-back WMMAs per wave.
// ---------------------------------------------------------------------------
__global__ __launch_bounds__(256) void k_dcn_gemm(
    const float* __restrict__ x, const _Float16* __restrict__ Whf,
    const float4* __restrict__ descW, const unsigned* __restrict__ descC,
    const float* __restrict__ bias, float* __restrict__ out)
{
  __shared__ _Float16 Wl[2][CO_ * 32];  // weight chunk (A), 2 x 16 KB
  __shared__ _Float16 S[64 * 32];       // sampled chunk (B), 4 KB
  __shared__ float4   dW[KK_ * 64];     // descriptor cache, 9 KB
  __shared__ unsigned dC[KK_ * 64];     // 2.25 KB

  int t  = threadIdx.x;
  int p0 = blockIdx.x * 64;
  int b  = p0 / HW, s0 = p0 % HW;
  const float* xb = x + b * CI_ * HW;

  for (int i = t; i < KK_ * 64; i += 256) {
    int kk = i >> 6, m = i & 63;
    int d  = (b * KK_ + kk) * HW + (s0 + m);
    dW[i] = descW[d];
    dC[i] = descC[d];
  }

  int lane = t & 31, wv = t >> 5;
  int nt     = wv & 3;
  int mtBase = (wv >> 2) * 8;
  int half = lane >> 4, ln = lane & 15;
  int mS = t >> 2;                      // producer: position 0..63
  int cq = (t & 3) * 8;                 // producer: channel sub-offset

  v8f zero = {};
  v8f acc[8];
  #pragma unroll
  for (int j = 0; j < 8; ++j) acc[j] = zero;

#if USE_TDM
  if (wv == 0)
    tdm_load_w_chunk((unsigned)(uintptr_t)&Wl[0][0], Whf, CO_);
#endif

  __syncthreads();                      // descriptors visible

  for (int kc = 0; kc < NCHUNK; ++kc) {
    int kk = kc >> 3, c0 = (kc & 7) << 5;
    int buf = kc & 1;

#if !USE_TDM
    v8h w0r, w1r, w2r, w3r;
    {
      const v8h* src = (const v8h*)(Whf + t * KDIM + kc * 32);
      w0r = src[0]; w1r = src[1]; w2r = src[2]; w3r = src[3];
    }
    if (kc + 1 < NCHUNK)
      __builtin_prefetch(Whf + t * KDIM + (kc + 1) * 32, 0, 1);
#endif

    // bilinear-sample 8 channels for position mS, tap kk
    v8h vals;
    {
      float4   wt = dW[kk * 64 + mS];
      unsigned pc = dC[kk * 64 + mS];
      int h0 = (int)(pc & 255u), h1 = (int)((pc >> 8) & 255u);
      int w0 = (int)((pc >> 16) & 255u), w1 = (int)(pc >> 24);
      int i00 = h0 * WW_ + w0, i01 = h0 * WW_ + w1;
      int i10 = h1 * WW_ + w0, i11 = h1 * WW_ + w1;
      const float* pch = xb + (c0 + cq) * HW;
      #pragma unroll
      for (int i = 0; i < 8; ++i) {
        const float* p = pch + i * HW;
        float v = wt.x * p[i00] + wt.y * p[i01] +
                  wt.z * p[i10] + wt.w * p[i11];
        vals[i] = (_Float16)v;
      }
    }

    __syncthreads();                    // prior chunk's fragment reads done
#if USE_TDM
    if (wv == 0 && kc + 1 < NCHUNK)     // DMA next weight panel into alt buf
      tdm_load_w_chunk((unsigned)(uintptr_t)&Wl[buf ^ 1][0],
                       Whf + (kc + 1) * 32, CO_);
#else
    *(v8h*)(&Wl[0][0] + t * 32 +  0) = w0r;
    *(v8h*)(&Wl[0][0] + t * 32 +  8) = w1r;
    *(v8h*)(&Wl[0][0] + t * 32 + 16) = w2r;
    *(v8h*)(&Wl[0][0] + t * 32 + 24) = w3r;
    buf = 0;
#endif
    *(v8h*)(S + mS * 32 + cq) = vals;
#if USE_TDM
    if (wv == 0) {                      // in-order TDM completion per wave
      if (kc + 1 < NCHUNK) __builtin_amdgcn_s_wait_tensorcnt(1);
      else                 __builtin_amdgcn_s_wait_tensorcnt(0);
    }
#endif
    __syncthreads();

    // B fragment: K {half*8..+7} and {16+half*8..+7}, matching A
    v8h blo = *(const v8h*)(S + (nt * 16 + ln) * 32 + half * 8);
    v8h bhi = *(const v8h*)(S + (nt * 16 + ln) * 32 + 16 + half * 8);
    v16h bf = cat16(blo, bhi);

    #pragma unroll
    for (int j = 0; j < 8; ++j) {
      int ch = (mtBase + j) * 16 + ln;
      // A fragment: K {half*8..+7} and {16+half*8..+7} (ISA 16-bit 16x32 A)
      v8h alo = *(const v8h*)(&Wl[buf][0] + ch * 32 + half * 8);
      v8h ahi = *(const v8h*)(&Wl[buf][0] + ch * 32 + 16 + half * 8);
      v16h af = cat16(alo, ahi);
      acc[j] = __builtin_amdgcn_wmma_f32_16x16x32_f16(false, af, false, bf,
                                                      (short)0, acc[j],
                                                      false, false);
    }
  }

  // epilogue: C tile — VGPR r => channel row r + half*8, lane ln => position
  int spos = s0 + nt * 16 + ln;
  #pragma unroll
  for (int j = 0; j < 8; ++j) {
    int chB = (mtBase + j) * 16 + half * 8;
    #pragma unroll
    for (int r = 0; r < 8; ++r) {
      int ch = chB + r;
      out[(size_t)(b * CO_ + ch) * HW + spos] = acc[j][r] + bias[ch];
    }
  }
}

// ---------------------------------------------------------------------------
// BatchNorm statistics: one block per channel.
// ---------------------------------------------------------------------------
__global__ __launch_bounds__(256) void k_bn_stats(
    const float* __restrict__ out, float* __restrict__ mean,
    float* __restrict__ var)
{
  __shared__ float rs[256], rq[256];
  int ch = blockIdx.x, t = threadIdx.x;
  float s = 0.0f, q = 0.0f;
  for (int i = t; i < NB * HW; i += 256) {
    int bb = i / HW, sp = i % HW;
    float v = out[(size_t)(bb * CO_ + ch) * HW + sp];
    s += v; q += v * v;
  }
  rs[t] = s; rq[t] = q;
  __syncthreads();
  for (int off = 128; off > 0; off >>= 1) {
    if (t < off) { rs[t] += rs[t + off]; rq[t] += rq[t + off]; }
    __syncthreads();
  }
  if (t == 0) {
    float mu = rs[0] * (1.0f / (NB * HW));
    mean[ch] = mu;
    var[ch]  = rq[0] * (1.0f / (NB * HW)) - mu * mu;
  }
}

__global__ __launch_bounds__(256) void k_bn_apply(
    float* __restrict__ out, const float* __restrict__ mean,
    const float* __restrict__ var, const float* __restrict__ gamma,
    const float* __restrict__ beta)
{
  int idx = blockIdx.x * 256 + threadIdx.x;   // NB*CO_*HW total
  int ch  = (idx / HW) % CO_;
  float v = out[idx];
  float y = (v - mean[ch]) * rsqrtf(var[ch] + 1e-5f) * gamma[ch] + beta[ch];
  out[idx] = fmaxf(y, 0.0f);
}

// ---------------------------------------------------------------------------
// Workspace layout (bytes, all 16-aligned):
//   om    : 0          B*27*HW f32      = 3,981,312
//   Whf   : 3,981,312  CO_*KDIM f16     = 1,179,648
//   Wohf  : 5,160,960  32*KDIM f16      =   147,456
//   descW : 5,308,416  331776 float4    = 5,308,416
//   descC : 10,616,832 331776 u32       = 1,327,104
//   stats : 11,943,936 512 f32          =     2,048
// ---------------------------------------------------------------------------
extern "C" void kernel_launch(void* const* d_in, const int* in_sizes, int n_in,
                              void* d_out, int out_size, void* d_ws, size_t ws_size,
                              hipStream_t stream) {
  const float* x     = (const float*)d_in[0];
  const float* w_off = (const float*)d_in[1];
  const float* b_off = (const float*)d_in[2];
  const float* w     = (const float*)d_in[3];
  const float* bias  = (const float*)d_in[4];
  const float* gamma = (const float*)d_in[5];
  const float* beta  = (const float*)d_in[6];
  float* out = (float*)d_out;

  char* ws = (char*)d_ws;
  float*    om    = (float*)(ws);
  _Float16* Whf   = (_Float16*)(ws + 3981312);
  _Float16* Wohf  = (_Float16*)(ws + 5160960);
  float4*   descW = (float4*)(ws + 5308416);
  unsigned* descC = (unsigned*)(ws + 10616832);
  float*    mean  = (float*)(ws + 11943936);
  float*    var   = mean + 256;

  k_cvt_weights<<<2592, 256, 0, stream>>>(w, w_off, Whf, Wohf);
  k_offset_conv<<<576, 256, 0, stream>>>(x, Wohf, b_off, om);
  k_prep<<<1296, 256, 0, stream>>>(om, descW, descC);
  k_dcn_gemm<<<576, 256, 0, stream>>>(x, Whf, descW, descC, bias, out);
  k_bn_stats<<<256, 256, 0, stream>>>(out, mean, var);
  k_bn_apply<<<36864, 256, 0, stream>>>(out, mean, var, gamma, beta);
}
